// MultiHeadSelfAttention_60722247631242
// MI455X (gfx1250) — matrix operand order repaired and checked by reading, not compiled, not run on hardware
//
#include <hip/hip_runtime.h>

typedef __bf16 bf16;
typedef __attribute__((ext_vector_type(8)))  bf16  v8bf;
typedef __attribute__((ext_vector_type(16))) bf16  v16bf;
typedef __attribute__((ext_vector_type(8)))  float v8f;

#define EMBED 1024
#define NHEAD 16
#define HDIM  64
#define SEQ   2048
#define BATCH 4
#define ROWS  (BATCH*SEQ)   /* 8192 */
#define QK_SCALE 0.125f     /* 64^-0.5 */

__device__ __forceinline__ v8bf ld8(const bf16* p) {
    v8bf v; __builtin_memcpy(&v, p, 16); return v;
}
__device__ __forceinline__ void st8(bf16* p, v8bf v) {
    __builtin_memcpy(p, &v, 16);
}
__device__ __forceinline__ v16bf cat16(v8bf lo, v8bf hi) {
    return __builtin_shufflevector(lo, hi, 0,1,2,3,4,5,6,7,8,9,10,11,12,13,14,15);
}

// ---------------------------------------------------------------- converters
__global__ void cvt_kernel(const float* __restrict__ x, bf16* __restrict__ y, int n) {
    int i = blockIdx.x * blockDim.x + threadIdx.x;
    int stride = gridDim.x * blockDim.x;
    for (; i < n; i += stride) y[i] = (bf16)x[i];
}

// w: [K][N] fp32 row-major  ->  wt: [N][K] bf16 row-major
__global__ void cvt_transpose_kernel(const float* __restrict__ w, bf16* __restrict__ wt,
                                     int K, int N) {
    long long i = (long long)blockIdx.x * blockDim.x + threadIdx.x;
    long long total = (long long)K * N;
    long long stride = (long long)gridDim.x * blockDim.x;
    for (; i < total; i += stride) {
        int k = (int)(i % K);
        int n = (int)(i / K);
        wt[(long long)n * K + k] = (bf16)w[(long long)k * N + n];
    }
}

// ---------------------------------------------------------------- GEMM
// C[row, col] = A[row, :] . Bt[col, :] + bias[col]
// Software-pipelined, double-buffered LDS (one barrier per BK step, global
// loads for step k+1 issued before the WMMAs of step k).
// MODE 0: QKV epilogue (scatter Q,K [B,H,N,D], V transposed [B,H,D,N], bf16, Q scaled)
// MODE 1: proj epilogue (fp32 output [ROWS][EMBED])
template <int MODE>
__global__ __launch_bounds__(256) void gemm_kernel(
    const bf16* __restrict__ A,    // [Mtot][1024]
    const bf16* __restrict__ Bt,   // [Ntot][1024]
    const float* __restrict__ bias,
    bf16* __restrict__ Qb, bf16* __restrict__ Kb, bf16* __restrict__ Vtb,
    float* __restrict__ out)
{
    constexpr int K = 1024;
    constexpr int NIT = K / 32;
    __shared__ bf16 As[2][128 * 40];   // 128 rows (M), 32 k-cols, stride 40 (pad)
    __shared__ bf16 Bs[2][128 * 40];   // 128 rows (N), 32 k-cols, stride 40

    const int t    = threadIdx.x;
    const int lane = t & 31;
    const int w    = t >> 5;
    const int wm   = w >> 2;      // 0..1  (64 rows each)
    const int wn   = w & 3;       // 0..3  (32 cols each)
    const int r    = lane & 15;
    const int ksel = lane >> 4;   // lane half
    const int rowBase = blockIdx.x * 128;
    const int colBase = blockIdx.y * 128;

    // per-thread staging coordinates: two 8-element chunks cover 128x32
    const int row0 = t >> 2,            col0 = (t & 3) << 3;
    const int row1 = (t + 256) >> 2,    col1 = ((t + 256) & 3) << 3;
    const bf16* Ag0 = A  + (size_t)(rowBase + row0) * K + col0;
    const bf16* Ag1 = A  + (size_t)(rowBase + row1) * K + col1;
    const bf16* Bg0 = Bt + (size_t)(colBase + row0) * K + col0;
    const bf16* Bg1 = Bt + (size_t)(colBase + row1) * K + col1;

    const v8f vzero = {};
    v8f acc[4][2];
    #pragma unroll
    for (int i = 0; i < 4; i++)
        #pragma unroll
        for (int j = 0; j < 2; j++) acc[i][j] = vzero;

    // prologue: fetch k-step 0 into registers
    v8bf ra0 = ld8(Ag0), ra1 = ld8(Ag1), rb0 = ld8(Bg0), rb1 = ld8(Bg1);

    for (int it = 0; it < NIT; ++it) {
        bf16* Ab = As[it & 1];
        bf16* Bb = Bs[it & 1];
        st8(&Ab[row0 * 40 + col0], ra0);
        st8(&Ab[row1 * 40 + col1], ra1);
        st8(&Bb[row0 * 40 + col0], rb0);
        st8(&Bb[row1 * 40 + col1], rb1);
        if (it + 1 < NIT) {
            int k0 = (it + 1) * 32;   // issue next chunk early; overlaps WMMAs
            ra0 = ld8(Ag0 + k0); ra1 = ld8(Ag1 + k0);
            rb0 = ld8(Bg0 + k0); rb1 = ld8(Bg1 + k0);
        }
        __syncthreads();

        v16bf af[4], bfm[2];
        #pragma unroll
        for (int mt = 0; mt < 4; mt++) {
            // A layout: lane=row, K interleave-8 between lane halves
            const bf16* p = &Ab[(wm * 64 + mt * 16 + r) * 40 + ksel * 8];
            af[mt] = cat16(ld8(p), ld8(p + 16));
        }
        #pragma unroll
        for (int nt = 0; nt < 2; nt++) {
            // B layout: lane=col, K interleave-8 between lane halves
            const bf16* p = &Bb[(wn * 32 + nt * 16 + r) * 40 + ksel * 8];
            bfm[nt] = cat16(ld8(p), ld8(p + 16));
        }
        #pragma unroll
        for (int mt = 0; mt < 4; mt++)
            #pragma unroll
            for (int nt = 0; nt < 2; nt++)
                acc[mt][nt] = __builtin_amdgcn_wmma_f32_16x16x32_bf16(
                    false, af[mt], false, bfm[nt], (short)0, acc[mt][nt], false, false);
    }

    // epilogue: C layout -> row = i + 8*laneHalf, col = lane&15
    if (MODE == 0) {
        // section is uniform per 128-wide column block (boundaries at 1024)
        const int sec = colBase >> 10;
        #pragma unroll
        for (int mt = 0; mt < 4; mt++)
        #pragma unroll
        for (int nt = 0; nt < 2; nt++)
        #pragma unroll
        for (int i = 0; i < 8; i++) {
            int row = rowBase + wm * 64 + mt * 16 + i + ksel * 8;
            int col = colBase + wn * 32 + nt * 16 + r;
            float v = acc[mt][nt][i] + bias[col];
            int cc = col & 1023;
            int h = cc >> 6, d = cc & 63;
            int bb = row >> 11, n = row & 2047;
            if (sec == 0) {
                Qb[((size_t)(bb * NHEAD + h) * SEQ + n) * HDIM + d] = (bf16)(v * QK_SCALE);
            } else if (sec == 1) {
                Kb[((size_t)(bb * NHEAD + h) * SEQ + n) * HDIM + d] = (bf16)v;
            } else {
                Vtb[((size_t)(bb * NHEAD + h) * HDIM + d) * SEQ + n] = (bf16)v;
            }
        }
    } else {
        #pragma unroll
        for (int mt = 0; mt < 4; mt++)
        #pragma unroll
        for (int nt = 0; nt < 2; nt++)
        #pragma unroll
        for (int i = 0; i < 8; i++) {
            int row = rowBase + wm * 64 + mt * 16 + i + ksel * 8;
            int col = colBase + wn * 32 + nt * 16 + r;
            out[(size_t)row * EMBED + col] = acc[mt][nt][i] + bias[col];
        }
    }
}

// ---------------------------------------------------------------- attention
// One workgroup (8 waves) = 128 query rows of one (b,h); waves share K/V chunks
// staged in double-buffered LDS (pipelined). Flash-attention over 64-key chunks.
// Q,K: [B*H][SEQ][HDIM] bf16 (Q pre-scaled). Vt: [B*H][HDIM][SEQ] bf16.
// Output AO: [B*SEQ][EMBED] bf16 (head-concatenated).
__global__ __launch_bounds__(256) void attn_kernel(
    const bf16* __restrict__ Qb, const bf16* __restrict__ Kb,
    const bf16* __restrict__ Vtb, bf16* __restrict__ AO)
{
    __shared__ bf16 Ks[2][64 * 72];   // [key][d], stride 72 (pad)
    __shared__ bf16 Vs[2][64 * 72];   // [d][key], stride 72
    __shared__ bf16 P[8][16 * 72];    // per-wave P tile (16x64), stride 72

    const int t = threadIdx.x, lane = t & 31, w = t >> 5;
    const int r = lane & 15, ksel = lane >> 4;
    const int bh   = blockIdx.x >> 4;        // 0..63
    const int qblk = blockIdx.x & 15;        // 0..15
    const int q0   = qblk * 128 + w * 16;
    const bf16* Qh = Qb  + (size_t)bh * SEQ * HDIM;
    const bf16* Kh = Kb  + (size_t)bh * SEQ * HDIM;
    const bf16* Vh = Vtb + (size_t)bh * HDIM * SEQ;

    // staging coords: each thread copies 2 K-chunks + 2 V-chunks of 8 bf16
    const int rowA = t >> 3,          colA = (t & 7) << 3;         // rows 0..31
    const int rowB = rowA + 32,       colB = colA;                 // rows 32..63

    const v8f vzero = {};
    v16bf aq[2];
    #pragma unroll
    for (int kk = 0; kk < 2; kk++) {
        const bf16* p = Qh + (size_t)(q0 + r) * HDIM + kk * 32 + ksel * 8;
        aq[kk] = cat16(ld8(p), ld8(p + 16));
    }

    v8f o[4];
    #pragma unroll
    for (int i = 0; i < 4; i++) o[i] = vzero;
    float mrow[8], lrow[8];
    #pragma unroll
    for (int i = 0; i < 8; i++) { mrow[i] = -1e30f; lrow[i] = 0.f; }

    bf16* Pw = &P[w][0];

    // prologue: fetch key-chunk 0 into registers
    v8bf gk0 = ld8(Kh + (size_t)rowA * HDIM + colA);
    v8bf gk1 = ld8(Kh + (size_t)rowB * HDIM + colB);
    v8bf gv0 = ld8(Vh + (size_t)rowA * SEQ + colA);
    v8bf gv1 = ld8(Vh + (size_t)rowB * SEQ + colB);

    for (int it = 0; it < SEQ / 64; ++it) {
        const int buf = it & 1;
        bf16* Kbuf = Ks[buf];
        bf16* Vbuf = Vs[buf];
        st8(&Kbuf[rowA * 72 + colA], gk0);
        st8(&Kbuf[rowB * 72 + colB], gk1);
        st8(&Vbuf[rowA * 72 + colA], gv0);
        st8(&Vbuf[rowB * 72 + colB], gv1);
        if (it + 1 < SEQ / 64) {
            int kvn = (it + 1) * 64;   // issue next chunk early; overlaps WMMAs
            gk0 = ld8(Kh + (size_t)(kvn + rowA) * HDIM + colA);
            gk1 = ld8(Kh + (size_t)(kvn + rowB) * HDIM + colB);
            gv0 = ld8(Vh + (size_t)rowA * SEQ + kvn + colA);
            gv1 = ld8(Vh + (size_t)rowB * SEQ + kvn + colB);
        }
        __syncthreads();

        // ---- S = Q K^T for 16x64 tile (fp32; 1/sqrt(D) pre-folded into Q)
        v8f s[4];
        #pragma unroll
        for (int nt = 0; nt < 4; nt++) {
            v8f z = vzero;
            #pragma unroll
            for (int kk = 0; kk < 2; kk++) {
                // B-frag: col = key row (row-major K == Bt), K-dim = d
                const bf16* p = &Kbuf[(nt * 16 + r) * 72 + kk * 32 + ksel * 8];
                v16bf bk = cat16(ld8(p), ld8(p + 16));
                z = __builtin_amdgcn_wmma_f32_16x16x32_bf16(
                        false, aq[kk], false, bk, (short)0, z, false, false);
            }
            s[nt] = z;
        }

        // ---- online softmax (rows live in 16-lane halves of the C layout)
        float cm[8];
        #pragma unroll
        for (int i = 0; i < 8; i++)
            cm[i] = fmaxf(fmaxf(s[0][i], s[1][i]), fmaxf(s[2][i], s[3][i]));
        #pragma unroll
        for (int off = 1; off < 16; off <<= 1)
            #pragma unroll
            for (int i = 0; i < 8; i++) cm[i] = fmaxf(cm[i], __shfl_xor(cm[i], off, 16));
        float scl[8];
        #pragma unroll
        for (int i = 0; i < 8; i++) {
            float nm = fmaxf(mrow[i], cm[i]);
            scl[i] = __expf(mrow[i] - nm);
            mrow[i] = nm;
        }
        float rs[8];
        #pragma unroll
        for (int i = 0; i < 8; i++) {
            float p0 = __expf(s[0][i] - mrow[i]);
            float p1 = __expf(s[1][i] - mrow[i]);
            float p2 = __expf(s[2][i] - mrow[i]);
            float p3 = __expf(s[3][i] - mrow[i]);
            s[0][i] = p0; s[1][i] = p1; s[2][i] = p2; s[3][i] = p3;
            rs[i] = (p0 + p1) + (p2 + p3);
        }
        #pragma unroll
        for (int off = 1; off < 16; off <<= 1)
            #pragma unroll
            for (int i = 0; i < 8; i++) rs[i] += __shfl_xor(rs[i], off, 16);
        #pragma unroll
        for (int i = 0; i < 8; i++) lrow[i] = lrow[i] * scl[i] + rs[i];
        #pragma unroll
        for (int dt = 0; dt < 4; dt++)
            #pragma unroll
            for (int i = 0; i < 8; i++) o[dt][i] = o[dt][i] * scl[i];

        // ---- re-layout P: C-fragments -> LDS (row-major 16x64 bf16) -> A-fragments
        #pragma unroll
        for (int nt = 0; nt < 4; nt++)
            #pragma unroll
            for (int i = 0; i < 8; i++) {
                int m = i + ksel * 8;
                Pw[m * 72 + nt * 16 + r] = (bf16)s[nt][i];
            }
        v16bf pa[2];
        #pragma unroll
        for (int kc = 0; kc < 2; kc++) {
            const bf16* pp = Pw + r * 72 + kc * 32 + ksel * 8;
            pa[kc] = cat16(ld8(pp), ld8(pp + 16));
        }

        // ---- O += P V   (B-frag: col = d, K-dim = keys; Vt-in-LDS row-major)
        #pragma unroll
        for (int dt = 0; dt < 4; dt++)
            #pragma unroll
            for (int kc = 0; kc < 2; kc++) {
                const bf16* vp = &Vbuf[(dt * 16 + r) * 72 + kc * 32 + ksel * 8];
                v16bf vb = cat16(ld8(vp), ld8(vp + 16));
                o[dt] = __builtin_amdgcn_wmma_f32_16x16x32_bf16(
                            false, pa[kc], false, vb, (short)0, o[dt], false, false);
            }
    }

    // ---- epilogue: normalize by lrow, store bf16 head-concatenated
    const int b = bh >> 4, h = bh & 15;
    #pragma unroll
    for (int dt = 0; dt < 4; dt++)
        #pragma unroll
        for (int i = 0; i < 8; i++) {
            int m = i + ksel * 8;
            float v = o[dt][i] / lrow[i];
            AO[((size_t)(b * SEQ + q0 + m)) * EMBED + h * HDIM + dt * 16 + r] = (bf16)v;
        }
}

// ---------------------------------------------------------------- launcher
extern "C" void kernel_launch(void* const* d_in, const int* in_sizes, int n_in,
                              void* d_out, int out_size, void* d_ws, size_t ws_size,
                              hipStream_t stream) {
    const float* x      = (const float*)d_in[0];
    const float* w_qkv  = (const float*)d_in[1];
    const float* b_qkv  = (const float*)d_in[2];
    const float* w_proj = (const float*)d_in[3];
    const float* b_proj = (const float*)d_in[4];
    float* out = (float*)d_out;

    char* ws = (char*)d_ws;
    size_t off = 0;
    auto alloc = [&](size_t bytes) -> void* {
        void* p = ws + off;
        off += (bytes + 255) & ~(size_t)255;
        return p;
    };
    bf16* xb     = (bf16*)alloc((size_t)ROWS * EMBED * 2);        // x in bf16
    bf16* wqkvT  = (bf16*)alloc((size_t)3 * EMBED * EMBED * 2);   // [3072][1024]
    bf16* wprojT = (bf16*)alloc((size_t)EMBED * EMBED * 2);       // [1024][1024]
    bf16* Qb     = (bf16*)alloc((size_t)ROWS * EMBED * 2);        // [B,H,N,D]
    bf16* Kb     = (bf16*)alloc((size_t)ROWS * EMBED * 2);        // [B,H,N,D]
    bf16* Vtb    = (bf16*)alloc((size_t)ROWS * EMBED * 2);        // [B,H,D,N]
    bf16* AO     = (bf16*)alloc((size_t)ROWS * EMBED * 2);        // [B*N][C]

    cvt_kernel<<<2048, 256, 0, stream>>>(x, xb, ROWS * EMBED);
    cvt_transpose_kernel<<<2048, 256, 0, stream>>>(w_qkv, wqkvT, EMBED, 3 * EMBED);
    cvt_transpose_kernel<<<1024, 256, 0, stream>>>(w_proj, wprojT, EMBED, EMBED);

    gemm_kernel<0><<<dim3(ROWS / 128, (3 * EMBED) / 128), 256, 0, stream>>>(
        xb, wqkvT, b_qkv, Qb, Kb, Vtb, nullptr);

    // 64 (b,h) pairs x 16 query-blocks of 128 rows
    attn_kernel<<<64 * 16, 256, 0, stream>>>(Qb, Kb, Vtb, AO);

    gemm_kernel<1><<<dim3(ROWS / 128, EMBED / 128), 256, 0, stream>>>(
        AO, wprojT, b_proj, nullptr, nullptr, nullptr, out);
}
